// MoTBlock_54700703482361
// MI455X (gfx1250) — matrix operand order repaired and checked by reading, not compiled, not run on hardware
//
#include <hip/hip_runtime.h>
#include <hip/hip_bf16.h>

typedef __bf16 bf16_t;
typedef __attribute__((ext_vector_type(4)))  float        fx4;
typedef __attribute__((ext_vector_type(8)))  float        v8f;
typedef __attribute__((ext_vector_type(4)))  __bf16       v4bf;
typedef __attribute__((ext_vector_type(8)))  __bf16       v8bf;
typedef __attribute__((ext_vector_type(16))) __bf16       v16bf;
typedef __attribute__((ext_vector_type(4)))  unsigned int v4u;
typedef __attribute__((ext_vector_type(4)))  int          v4i;
typedef __attribute__((ext_vector_type(8)))  int          v8i;

#define NTOK   4096
#define DMODEL 1024
#define TD     3072
#define FFDIM  4096
#define SEQ    1024
#define NHEAD  16

__device__ __forceinline__ v8f wmma_bf16f32(v16bf a, v16bf b, v8f c) {
  // D = A(16x32 bf16) x B(32x16 bf16) + C(16x16 f32)
  return __builtin_amdgcn_wmma_f32_16x16x32_bf16(false, a, false, b, (short)0, c,
                                                 false, false);
}

__device__ __forceinline__ v16bf ld_frag2(const bf16_t* lo, const bf16_t* hi) {
  v8bf a = *(const v8bf*)lo;
  v8bf b = *(const v8bf*)hi;
  v16bf r;
#pragma unroll
  for (int i = 0; i < 8; ++i) { r[i] = a[i]; r[i + 8] = b[i]; }
  return r;
}

__device__ __forceinline__ v8f zero_v8f() {
  v8f z;
#pragma unroll
  for (int i = 0; i < 8; ++i) z[i] = 0.0f;
  return z;
}

__device__ __forceinline__ float gelu_tanh(float x) {
  float u = 0.7978845608028654f * (x + 0.044715f * x * x * x);
  return 0.5f * x * (1.0f + tanhf(u));
}

// Issue a TDM 2D tile load: Wt tile [tileN=128 rows][tileK=32 bf16] at
// global ga (row stride KT elements) -> LDS ldsOff with 80B row pitch
// (pad_interval = 16 DWORDs data, pad_amount = 4 DWORDs).  ISA 8.3/8.4.
__device__ __forceinline__ void tdm_load_tile_b(unsigned ldsOff,
                                                const bf16_t* gptr, int KT) {
  unsigned long long ga = (unsigned long long)(uintptr_t)gptr;
  v4u g0;
  g0[0] = 1u;  // count=1, user mode, no gather
  g0[1] = ldsOff;
  g0[2] = (unsigned)(ga & 0xFFFFFFFFu);
  g0[3] = (unsigned)((ga >> 32) & 0x1FFFFFFu) | (2u << 30);  // type=2 (image)
  v8i g1;
  // data_size=1 (2B) | pad_enable | pad_interval=3 (16 dw) | pad_amount=3 (4 dw)
  g1[0] = (int)((1u << 16) | (1u << 20) | (3u << 22) | (3u << 25));
  unsigned td0 = (unsigned)KT;      // tensor dim0 (k) in elements
  unsigned td1 = (1u << 20);        // tensor dim1 (rows), generous bound
  g1[1] = (int)((td0 & 0xFFFFu) << 16);
  g1[2] = (int)(((td0 >> 16) & 0xFFFFu) | ((td1 & 0xFFFFu) << 16));
  g1[3] = (int)(((td1 >> 16) & 0xFFFFu) | (32u << 16));  // tile_dim0 = 32
  g1[4] = (int)(128u);                                   // tile_dim1 = 128
  unsigned long long s0 = (unsigned long long)KT;        // dim0 stride (elems)
  g1[5] = (int)(s0 & 0xFFFFFFFFu);
  g1[6] = (int)((s0 >> 32) & 0xFFFFu);
  g1[7] = 0;
  v4i g2; g2[0] = g2[1] = g2[2] = g2[3] = 0;
  v4i g3; g3[0] = g3[1] = g3[2] = g3[3] = 0;
#if defined(__clang_major__) && (__clang_major__ >= 23)
  v8i g4;
#pragma unroll
  for (int i = 0; i < 8; ++i) g4[i] = 0;
  __builtin_amdgcn_tensor_load_to_lds(g0, g1, g2, g3, g4, 0);
#else
  __builtin_amdgcn_tensor_load_to_lds(g0, g1, g2, g3, 0);
#endif
}

// ---------------------------------------------------------------------------
// Weight convert + transpose: W[K,N] f32 -> Wt[N,K] bf16 (32x32 tiles).
// ---------------------------------------------------------------------------
__global__ __launch_bounds__(256) void transpose_bf16_kernel(
    const float* __restrict__ W, bf16_t* __restrict__ Wt, int K, int N) {
  __shared__ float tile[32][33];
  const int t = threadIdx.x;
  const int nb = blockIdx.x * 32;
  const int kb = blockIdx.y * 32;
  const int r = t >> 5;  // 0..7
  const int c = t & 31;
#pragma unroll
  for (int i = 0; i < 4; ++i)
    tile[r + 8 * i][c] = W[(size_t)(kb + r + 8 * i) * N + nb + c];
  __syncthreads();
  const int nr = t >> 3;       // output row (n) 0..31
  const int kc = (t & 7) * 4;  // 4 contiguous k
  v4bf o;
#pragma unroll
  for (int j = 0; j < 4; ++j) o[j] = (bf16_t)tile[kc + j][nr];
  *(v4bf*)&Wt[(size_t)(nb + nr) * K + kb + kc] = o;
}

// ---------------------------------------------------------------------------
// LayerNorm (normalize only) + router softmax.  One block per token.
// ---------------------------------------------------------------------------
__global__ __launch_bounds__(256) void ln_router_kernel(
    const float* __restrict__ x, const float* __restrict__ Wr,
    const float* __restrict__ br, bf16_t* __restrict__ hn,
    float* __restrict__ score) {
  __shared__ float red[8];
  __shared__ float red4[8][4];
  __shared__ float sStat[2];
  const int n = blockIdx.x;
  const int t = threadIdx.x;
  const int wave = t >> 5, lane = t & 31;

  fx4 v = *(const fx4*)(x + (size_t)n * DMODEL + t * 4);
  float s = v[0] + v[1] + v[2] + v[3];
#pragma unroll
  for (int m = 1; m < 32; m <<= 1) s += __shfl_xor(s, m, 32);
  if (lane == 0) red[wave] = s;
  __syncthreads();
  if (t == 0) {
    float tot = 0.f;
#pragma unroll
    for (int i = 0; i < 8; ++i) tot += red[i];
    sStat[0] = tot * (1.0f / DMODEL);
  }
  __syncthreads();
  const float mu = sStat[0];
  float d0 = v[0] - mu, d1 = v[1] - mu, d2 = v[2] - mu, d3 = v[3] - mu;
  float sq = d0 * d0 + d1 * d1 + d2 * d2 + d3 * d3;
#pragma unroll
  for (int m = 1; m < 32; m <<= 1) sq += __shfl_xor(sq, m, 32);
  if (lane == 0) red[wave] = sq;
  __syncthreads();
  if (t == 0) {
    float tot = 0.f;
#pragma unroll
    for (int i = 0; i < 8; ++i) tot += red[i];
    sStat[1] = rsqrtf(tot * (1.0f / DMODEL) + 1e-5f);
  }
  // router logits from RAW h (reference: softmax(h @ Wr + br))
  float le[4] = {0.f, 0.f, 0.f, 0.f};
#pragma unroll
  for (int i = 0; i < 4; ++i) {
    fx4 w = *(const fx4*)(Wr + (size_t)(t * 4 + i) * 4);
#pragma unroll
    for (int e = 0; e < 4; ++e) le[e] += v[i] * w[e];
  }
#pragma unroll
  for (int e = 0; e < 4; ++e) {
#pragma unroll
    for (int m = 1; m < 32; m <<= 1) le[e] += __shfl_xor(le[e], m, 32);
  }
  if (lane == 0) {
#pragma unroll
    for (int e = 0; e < 4; ++e) red4[wave][e] = le[e];
  }
  __syncthreads();
  if (t == 0) {
    float lg[4];
#pragma unroll
    for (int e = 0; e < 4; ++e) {
      float tot = br[e];
#pragma unroll
      for (int i = 0; i < 8; ++i) tot += red4[i][e];
      lg[e] = tot;
    }
    float mx = fmaxf(fmaxf(lg[0], lg[1]), fmaxf(lg[2], lg[3]));
    float den = 0.f;
#pragma unroll
    for (int e = 0; e < 4; ++e) { lg[e] = __expf(lg[e] - mx); den += lg[e]; }
    float inv = 1.0f / den;
#pragma unroll
    for (int e = 0; e < 4; ++e) score[(size_t)n * 4 + e] = lg[e] * inv;
  }
  __syncthreads();
  const float rinv = sStat[1];
  v4bf hv;
#pragma unroll
  for (int i = 0; i < 4; ++i) hv[i] = (bf16_t)((v[i] - mu) * rinv);
  *(v4bf*)(hn + (size_t)n * DMODEL + t * 4) = hv;
}

// ---------------------------------------------------------------------------
// Routed GEMM (score folded into A rows -> plain WMMA GEMM).
// Block = 128x128 tile, 8 waves, each wave 64x32 (4x2 WMMA tiles).
// Double-buffered software pipeline: tile ks+1 staged (TDM for B, VALU for A)
// while tile ks is consumed by WMMA; one barrier per k-step.
// ---------------------------------------------------------------------------
enum { AM_PLAIN = 0, AM_AFFINE = 1, AM_SCORE = 2, AM_AFFSCORE = 3 };
enum { EP_QKV = 0, EP_PROJ = 1, EP_FC = 2, EP_P2 = 3 };

template <int AMODE, int EPI, bool USE_TDM>
__global__ __launch_bounds__(256) void moe_gemm_kernel(
    const bf16_t* __restrict__ Abase, const float* __restrict__ score,
    const float* __restrict__ gvec, const float* __restrict__ bvec,
    const float* __restrict__ W, const bf16_t* __restrict__ Wt, int KT,
    int kOff, const float* __restrict__ bias, const float* __restrict__ resid,
    float* __restrict__ outF, bf16_t* __restrict__ outB, int K, int Astride,
    int Ncols, int expert) {
  __shared__ bf16_t As[2][128 * 40];  // A' tile, [row][k], pitch 40
  __shared__ bf16_t Bs[2][128 * 40];  // B tile transposed, [col][k], pitch 40
  const int t = threadIdx.x;
  const int rowBase = blockIdx.y * 128;
  const int colBase = blockIdx.x * 128;
  const int wave = t >> 5, lane = t & 31;
  const int wm = wave & 1, wn = wave >> 1;
  const int half = lane >> 4, l16 = lane & 15;

  v8f acc[4][2];
#pragma unroll
  for (int mt = 0; mt < 4; ++mt)
#pragma unroll
    for (int nt = 0; nt < 2; ++nt) acc[mt][nt] = zero_v8f();

  const int nk = K >> 5;
  const int ar = t >> 3;         // A stage: row group 0..31
  const int ac = (t & 7) * 4;    // A stage: 4 contiguous k
  const int brw = t >> 3;        // B stage (fallback): k row 0..31
  const int bcl = (t & 7) * 16;  // B stage (fallback): 16 contiguous cols
  const unsigned bsOff[2] = {(unsigned)(uintptr_t)&Bs[0][0],
                             (unsigned)(uintptr_t)&Bs[1][0]};

  // ---- staging helpers --------------------------------------------------
  auto stage_a = [&](int ks, bf16_t* dst) {
    const int kk = (ks << 5) + ac;
    const int e = kk >> 10;            // expert (K = E*1024 paths)
    const int d = kk & (Astride - 1);  // Astride is a power of two
#pragma unroll
    for (int i = 0; i < 4; ++i) {
      const int row = ar + 32 * i;
      const int n = rowBase + row;
      const bf16_t* ap = Abase + (size_t)n * Astride + d;
      float vals[4];
      if (AMODE == AM_PLAIN) {
#pragma unroll
        for (int j = 0; j < 4; ++j) vals[j] = (float)ap[j];
      } else if (AMODE == AM_AFFINE) {
        const float* gp = gvec + (size_t)expert * Astride + d;
        const float* bp = bvec + (size_t)expert * Astride + d;
#pragma unroll
        for (int j = 0; j < 4; ++j) vals[j] = (float)ap[j] * gp[j] + bp[j];
      } else if (AMODE == AM_SCORE) {
        const float sc = score[(size_t)n * 4 + e];
#pragma unroll
        for (int j = 0; j < 4; ++j) vals[j] = sc * (float)ap[j];
      } else {  // AM_AFFSCORE
        const float sc = score[(size_t)n * 4 + e];
        const float* gp = gvec + (size_t)e * 1024 + d;
        const float* bp = bvec + (size_t)e * 1024 + d;
#pragma unroll
        for (int j = 0; j < 4; ++j)
          vals[j] = sc * ((float)ap[j] * gp[j] + bp[j]);
      }
      v4bf hv;
#pragma unroll
      for (int j = 0; j < 4; ++j) hv[j] = (bf16_t)vals[j];
      *(v4bf*)&dst[row * 40 + ac] = hv;
    }
  };
  auto stage_b_manual = [&](int ks, bf16_t* dst) {
    const float* wp = W + (size_t)((ks << 5) + brw) * Ncols + colBase + bcl;
#pragma unroll
    for (int q = 0; q < 4; ++q) {
      fx4 w = *(const fx4*)(wp + q * 4);
#pragma unroll
      for (int j = 0; j < 4; ++j)
        dst[(bcl + q * 4 + j) * 40 + brw] = (bf16_t)w[j];
    }
    __builtin_prefetch(wp + (size_t)32 * Ncols, 0, 1);
  };

  // ---- prologue: stage tile 0 into buffer 0 ------------------------------
  if (USE_TDM) {
    if (t < 32)
      tdm_load_tile_b(bsOff[0], Wt + (size_t)colBase * KT + kOff, KT);
  }
  stage_a(0, &As[0][0]);
  if (!USE_TDM)
    stage_b_manual(0, &Bs[0][0]);
  else if (t < 32)
    __builtin_amdgcn_s_wait_tensorcnt(0);
  __syncthreads();

  // ---- pipelined main loop ------------------------------------------------
  for (int ks = 0; ks < nk; ++ks) {
    const int cur = ks & 1;
    const int nxt = cur ^ 1;
    const bool have_next = (ks + 1 < nk);
    if (have_next) {
      if (USE_TDM) {
        if (t < 32)
          tdm_load_tile_b(bsOff[nxt],
                          Wt + (size_t)colBase * KT + kOff + ((ks + 1) << 5),
                          KT);
      }
      stage_a(ks + 1, &As[nxt][0]);
      if (!USE_TDM) stage_b_manual(ks + 1, &Bs[nxt][0]);
    }
    // ---- WMMA compute on current buffers (overlaps with staging/DMA) ----
    const bf16_t* Ac = &As[cur][0];
    const bf16_t* Bc = &Bs[cur][0];
    v16bf af[4], bfrag[2];
#pragma unroll
    for (int mt = 0; mt < 4; ++mt) {
      const int m = wm * 64 + mt * 16 + l16;
      af[mt] = ld_frag2(&Ac[m * 40 + half * 8], &Ac[m * 40 + 16 + half * 8]);
    }
#pragma unroll
    for (int nt = 0; nt < 2; ++nt) {
      const int c = wn * 32 + nt * 16 + l16;
      bfrag[nt] =
          ld_frag2(&Bc[c * 40 + half * 8], &Bc[c * 40 + 16 + half * 8]);
    }
#pragma unroll
    for (int mt = 0; mt < 4; ++mt)
#pragma unroll
      for (int nt = 0; nt < 2; ++nt)
        acc[mt][nt] = wmma_bf16f32(af[mt], bfrag[nt], acc[mt][nt]);
    if (USE_TDM && have_next && t < 32)
      __builtin_amdgcn_s_wait_tensorcnt(0);  // drain DMA for tile ks+1
    __syncthreads();
  }

  // ---- epilogue ----
#pragma unroll
  for (int nt = 0; nt < 2; ++nt) {
    const int col = colBase + wn * 32 + nt * 16 + l16;
    float bb[4] = {0.f, 0.f, 0.f, 0.f};
    if (EPI == EP_QKV || EPI == EP_PROJ) {
#pragma unroll
      for (int e = 0; e < 4; ++e) bb[e] = bias[(size_t)e * Ncols + col];
    }
    float bcol = 0.f;
    if (EPI == EP_FC) bcol = bias[(size_t)expert * Ncols + col];
#pragma unroll
    for (int mt = 0; mt < 4; ++mt) {
#pragma unroll
      for (int r = 0; r < 8; ++r) {
        const int row = rowBase + wm * 64 + mt * 16 + r + 8 * half;
        float v = acc[mt][nt][r];
        const size_t oi = (size_t)row * Ncols + col;
        if (EPI == EP_QKV) {
          fx4 sc = *(const fx4*)(score + (size_t)row * 4);
          v += sc[0] * bb[0] + sc[1] * bb[1] + sc[2] * bb[2] + sc[3] * bb[3];
          outB[oi] = (bf16_t)v;
        } else if (EPI == EP_PROJ) {
          fx4 sc = *(const fx4*)(score + (size_t)row * 4);
          v += sc[0] * bb[0] + sc[1] * bb[1] + sc[2] * bb[2] + sc[3] * bb[3];
          outF[oi] = v + resid[oi];
        } else if (EPI == EP_FC) {
          float g = gelu_tanh(v + bcol);
          outB[oi] = (bf16_t)(g * score[(size_t)row * 4 + expert]);
        } else {  // EP_P2: accumulate expert contribution
          outF[oi] += v;
        }
      }
    }
  }
}

// ---------------------------------------------------------------------------
// Flash-style causal attention: block = 128 q-rows of one (b,h), 8 waves,
// each wave owns 16 q-rows; K/V tiles (64 keys) staged in LDS as bf16.
// ---------------------------------------------------------------------------
__global__ __launch_bounds__(256) void attn_kernel(const bf16_t* __restrict__ qkv,
                                                   bf16_t* __restrict__ aout) {
  __shared__ bf16_t Ks[64 * 72];     // [key][hd], pitch 72
  __shared__ bf16_t Vt[64 * 72];     // [hd][key], pitch 72
  __shared__ bf16_t Ps[8][16 * 72];  // per-wave P tile [qrow][key]
  const int t = threadIdx.x;
  const int wave = t >> 5, lane = t & 31;
  const int half = lane >> 4, l16 = lane & 15;
  const int bh = blockIdx.y;
  const int b = bh >> 4, h = bh & 15;
  const int qBlock = blockIdx.x * 128;
  const int q0 = qBlock + wave * 16;
  const size_t base = (size_t)b * SEQ * TD;

  v16bf qf[2];
#pragma unroll
  for (int kc = 0; kc < 2; ++kc) {
    const bf16_t* qp =
        qkv + base + (size_t)(q0 + l16) * TD + h * 64 + kc * 32 + half * 8;
    qf[kc] = ld_frag2(qp, qp + 16);
  }
  v8f o[4];
#pragma unroll
  for (int od = 0; od < 4; ++od) o[od] = zero_v8f();
  float mrow[8], lrow[8];
#pragma unroll
  for (int r = 0; r < 8; ++r) { mrow[r] = -3.0e38f; lrow[r] = 0.f; }

  const int ktn = qBlock / 64 + 2;
  for (int kt = 0; kt < ktn; ++kt) {
    const int kb = kt * 64;
    {  // stage K row-major, V transposed
      const int key = t >> 2;
      const int d0 = (t & 3) * 16;
      const bf16_t* kp =
          qkv + base + (size_t)(kb + key) * TD + DMODEL + h * 64 + d0;
      *(v8bf*)&Ks[key * 72 + d0] = *(const v8bf*)kp;
      *(v8bf*)&Ks[key * 72 + d0 + 8] = *(const v8bf*)(kp + 8);
      const bf16_t* vp =
          qkv + base + (size_t)(kb + key) * TD + 2 * DMODEL + h * 64 + d0;
#pragma unroll
      for (int i = 0; i < 16; ++i) Vt[(d0 + i) * 72 + key] = vp[i];
    }
    __syncthreads();
    // S = Q @ K^T  (16 x 64)
    v8f sa[4];
#pragma unroll
    for (int nt = 0; nt < 4; ++nt) sa[nt] = zero_v8f();
#pragma unroll
    for (int nt = 0; nt < 4; ++nt) {
#pragma unroll
      for (int kc = 0; kc < 2; ++kc) {
        const bf16_t* bp = &Ks[(nt * 16 + l16) * 72 + kc * 32 + half * 8];
        v16bf kf = ld_frag2(bp, bp + 16);
        sa[nt] = wmma_bf16f32(qf[kc], kf, sa[nt]);
      }
    }
    // online softmax (row state replicated over each 16-lane group)
#pragma unroll
    for (int r = 0; r < 8; ++r) {
      const int rq = q0 + r + 8 * half;
      float sv[4];
      float rmax = -3.0e38f;
#pragma unroll
      for (int nt = 0; nt < 4; ++nt) {
        const int col = kb + nt * 16 + l16;
        float s = sa[nt][r] * 0.125f;  // 1/sqrt(64)
        s = (col <= rq) ? s : -3.0e38f;
        sv[nt] = s;
        rmax = fmaxf(rmax, s);
      }
#pragma unroll
      for (int m = 1; m < 16; m <<= 1)
        rmax = fmaxf(rmax, __shfl_xor(rmax, m, 32));
      const float newm = fmaxf(mrow[r], rmax);
      const float alpha = __expf(mrow[r] - newm);
      float rs = 0.f;
#pragma unroll
      for (int nt = 0; nt < 4; ++nt) {
        const float p = __expf(sv[nt] - newm);
        rs += p;
        Ps[wave][(r + 8 * half) * 72 + nt * 16 + l16] = (bf16_t)p;
      }
#pragma unroll
      for (int m = 1; m < 16; m <<= 1) rs += __shfl_xor(rs, m, 32);
      lrow[r] = lrow[r] * alpha + rs;
      mrow[r] = newm;
#pragma unroll
      for (int od = 0; od < 4; ++od) o[od][r] *= alpha;
    }
    // O += P @ V   (same-wave LDS RAW on Ps: DScnt ordering)
#pragma unroll
    for (int kc = 0; kc < 2; ++kc) {
      const bf16_t* pp = &Ps[wave][l16 * 72 + kc * 32 + half * 8];
      v16bf pf = ld_frag2(pp, pp + 16);
#pragma unroll
      for (int od = 0; od < 4; ++od) {
        const bf16_t* vb = &Vt[(od * 16 + l16) * 72 + kc * 32 + half * 8];
        v16bf vf = ld_frag2(vb, vb + 16);
        o[od] = wmma_bf16f32(pf, vf, o[od]);
      }
    }
    __syncthreads();
  }
#pragma unroll
  for (int od = 0; od < 4; ++od) {
#pragma unroll
    for (int r = 0; r < 8; ++r) {
      const int rq = q0 + r + 8 * half;
      const float val = o[od][r] / lrow[r];
      aout[(size_t)(b * SEQ + rq) * DMODEL + h * 64 + od * 16 + l16] =
          (bf16_t)val;
    }
  }
}

// out = y + score2 @ bp2   (p2 GEMMs then accumulate expert terms on top)
__global__ __launch_bounds__(256) void init_out_kernel(
    const float* __restrict__ y, const float* __restrict__ score,
    const float* __restrict__ bp2, float* __restrict__ out) {
  const int idx = blockIdx.x * 256 + threadIdx.x;
  const int n = idx >> 10;
  const int d = idx & 1023;
  const fx4 sc = *(const fx4*)(score + (size_t)n * 4);
  out[idx] = y[idx] + sc[0] * bp2[d] + sc[1] * bp2[1024 + d] +
             sc[2] * bp2[2048 + d] + sc[3] * bp2[3072 + d];
}

// ---------------------------------------------------------------------------
extern "C" void kernel_launch(void* const* d_in, const int* in_sizes, int n_in,
                              void* d_out, int out_size, void* d_ws,
                              size_t ws_size, hipStream_t stream) {
  (void)in_sizes; (void)n_in; (void)out_size;
  const float* x    = (const float*)d_in[0];
  const float* Wr_a = (const float*)d_in[1];
  const float* br_a = (const float*)d_in[2];
  const float* g_a  = (const float*)d_in[3];
  const float* b_a  = (const float*)d_in[4];
  const float* Wqkv = (const float*)d_in[5];
  const float* bqkv = (const float*)d_in[6];
  const float* Wpr  = (const float*)d_in[7];
  const float* bpr  = (const float*)d_in[8];
  const float* Wr_m = (const float*)d_in[9];
  const float* br_m = (const float*)d_in[10];
  const float* g_m  = (const float*)d_in[11];
  const float* b_m  = (const float*)d_in[12];
  const float* Wfc  = (const float*)d_in[13];
  const float* bfc  = (const float*)d_in[14];
  const float* Wp2  = (const float*)d_in[15];
  const float* bp2  = (const float*)d_in[16];
  float* out = (float*)d_out;

  char* ws = (char*)d_ws;
  size_t off = 0;
  auto take = [&](size_t bytes) -> char* {
    char* p = ws + off;
    off = (off + bytes + 255) & ~(size_t)255;
    return p;
  };
  bf16_t* hn_a  = (bf16_t*)take((size_t)NTOK * DMODEL * 2);
  float*  sc_a  = (float*) take((size_t)NTOK * 4 * 4);
  bf16_t* qkvB  = (bf16_t*)take((size_t)NTOK * TD * 2);
  bf16_t* attnA = (bf16_t*)take((size_t)NTOK * DMODEL * 2);
  float*  yBuf  = (float*) take((size_t)NTOK * DMODEL * 4);
  bf16_t* hn_m  = (bf16_t*)take((size_t)NTOK * DMODEL * 2);
  float*  sc_m  = (float*) take((size_t)NTOK * 4 * 4);
  bf16_t* inter = (bf16_t*)take((size_t)NTOK * FFDIM * 2);
  // pre-transposed bf16 weights (TDM path)
  bf16_t* WqkvT = (bf16_t*)take((size_t)TD * 4 * DMODEL * 2);      // [3072][4096]
  bf16_t* WprT  = (bf16_t*)take((size_t)DMODEL * 4 * DMODEL * 2);  // [1024][4096]
  bf16_t* WfcT  = (bf16_t*)take((size_t)FFDIM * 4 * DMODEL * 2);   // [4096][4096]
  bf16_t* Wp2T  = (bf16_t*)take((size_t)DMODEL * 4 * FFDIM * 2);   // [1024][16384]
  const bool use_tdm = (ws_size >= off);

  if (use_tdm) {
    transpose_bf16_kernel<<<dim3(TD / 32, (4 * DMODEL) / 32), 256, 0, stream>>>(
        Wqkv, WqkvT, 4 * DMODEL, TD);
    transpose_bf16_kernel<<<dim3(DMODEL / 32, (4 * DMODEL) / 32), 256, 0,
                            stream>>>(Wpr, WprT, 4 * DMODEL, DMODEL);
    transpose_bf16_kernel<<<dim3(FFDIM / 32, (4 * DMODEL) / 32), 256, 0,
                            stream>>>(Wfc, WfcT, 4 * DMODEL, FFDIM);
    transpose_bf16_kernel<<<dim3(DMODEL / 32, (4 * FFDIM) / 32), 256, 0,
                            stream>>>(Wp2, Wp2T, 4 * FFDIM, DMODEL);
  }

  // 1) attention router + LN
  ln_router_kernel<<<NTOK, 256, 0, stream>>>(x, Wr_a, br_a, hn_a, sc_a);
  // 2) routed QKV as one GEMM with K = E*D (score/affine folded into A)
  if (use_tdm)
    moe_gemm_kernel<AM_AFFSCORE, EP_QKV, true>
        <<<dim3(TD / 128, NTOK / 128), 256, 0, stream>>>(
            hn_a, sc_a, g_a, b_a, nullptr, WqkvT, 4 * DMODEL, 0, bqkv, nullptr,
            nullptr, qkvB, 4 * DMODEL, DMODEL, TD, 0);
  else
    moe_gemm_kernel<AM_AFFSCORE, EP_QKV, false>
        <<<dim3(TD / 128, NTOK / 128), 256, 0, stream>>>(
            hn_a, sc_a, g_a, b_a, Wqkv, nullptr, 0, 0, bqkv, nullptr, nullptr,
            qkvB, 4 * DMODEL, DMODEL, TD, 0);
  // 3) causal flash attention
  attn_kernel<<<dim3(SEQ / 128, 4 * NHEAD), 256, 0, stream>>>(qkvB, attnA);
  // 4) routed projection + residual -> y
  if (use_tdm)
    moe_gemm_kernel<AM_SCORE, EP_PROJ, true>
        <<<dim3(DMODEL / 128, NTOK / 128), 256, 0, stream>>>(
            attnA, sc_a, nullptr, nullptr, nullptr, WprT, 4 * DMODEL, 0, bpr, x,
            yBuf, nullptr, 4 * DMODEL, DMODEL, DMODEL, 0);
  else
    moe_gemm_kernel<AM_SCORE, EP_PROJ, false>
        <<<dim3(DMODEL / 128, NTOK / 128), 256, 0, stream>>>(
            attnA, sc_a, nullptr, nullptr, Wpr, nullptr, 0, 0, bpr, x, yBuf,
            nullptr, 4 * DMODEL, DMODEL, DMODEL, 0);
  // 5) MLP router + LN
  ln_router_kernel<<<NTOK, 256, 0, stream>>>(yBuf, Wr_m, br_m, hn_m, sc_m);
  // 6) out = y + score2 @ bp2
  init_out_kernel<<<(NTOK * DMODEL) / 256, 256, 0, stream>>>(yBuf, sc_m, bp2,
                                                             out);
  // 7) per-expert MLP: inter' = score2 * gelu(fc); out += inter' @ Wp2_e
  for (int e = 0; e < 4; ++e) {
    if (use_tdm) {
      moe_gemm_kernel<AM_AFFINE, EP_FC, true>
          <<<dim3(FFDIM / 128, NTOK / 128), 256, 0, stream>>>(
              hn_m, sc_m, g_m, b_m, nullptr, WfcT, 4 * DMODEL, e * DMODEL, bfc,
              nullptr, nullptr, inter, DMODEL, DMODEL, FFDIM, e);
      moe_gemm_kernel<AM_PLAIN, EP_P2, true>
          <<<dim3(DMODEL / 128, NTOK / 128), 256, 0, stream>>>(
              inter, nullptr, nullptr, nullptr, nullptr, Wp2T, 4 * FFDIM,
              e * FFDIM, nullptr, nullptr, out, nullptr, FFDIM, FFDIM, DMODEL,
              e);
    } else {
      moe_gemm_kernel<AM_AFFINE, EP_FC, false>
          <<<dim3(FFDIM / 128, NTOK / 128), 256, 0, stream>>>(
              hn_m, sc_m, g_m, b_m, Wfc + (size_t)e * DMODEL * FFDIM, nullptr,
              0, 0, bfc, nullptr, nullptr, inter, DMODEL, DMODEL, FFDIM, e);
      moe_gemm_kernel<AM_PLAIN, EP_P2, false>
          <<<dim3(DMODEL / 128, NTOK / 128), 256, 0, stream>>>(
              inter, nullptr, nullptr, nullptr,
              Wp2 + (size_t)e * FFDIM * DMODEL, nullptr, 0, 0, nullptr,
              nullptr, out, nullptr, FFDIM, FFDIM, DMODEL, e);
    }
  }
}
